// Subset_62311385530393
// MI455X (gfx1250) — compile-verified, matrix operand order checked by reading, not run on hardware
//
#include <hip/hip_runtime.h>
#include <hip/hip_bf16.h>
#include <math.h>

// Problem constants (fixed by setup_inputs)
#define B_  4
#define M_  4096
#define C_  256
#define N_  2048
#define STDDEV_F (4.0f / 4096.0f)          // 0.0009765625
#define INV_F    724.0773439f              // 1/(sqrt(2)*stddev)

typedef __attribute__((ext_vector_type(16))) _Float16 v16h;
typedef __attribute__((ext_vector_type(8)))  float    v8f;

// ---------------------------------------------------------------------------
// Branch-free 0.5 + 0.5*erf(t).  A&S 7.1.26 (|err| < 1.5e-7), with hardware
// v_rcp_f32 and v_exp_f32 (no ocml control flow).  f32 erf(z) == 1.0 for
// z >= 3.9192, so clamping there is exact.
// ---------------------------------------------------------------------------
__device__ __forceinline__ float cdf_fast(float t) {
    float z  = fminf(fabsf(t), 3.9192f);
    float k  = __builtin_amdgcn_rcpf(fmaf(0.3275911f, z, 1.0f));
    float p  = fmaf(1.061405429f, k, -1.453152027f);
    p = fmaf(p, k,  1.421413741f);
    p = fmaf(p, k, -0.284496736f);
    p = fmaf(p, k,  0.254829592f);
    p = p * k;
    float ex = __builtin_amdgcn_exp2f(-(z * z) * 1.44269504f);
    float e  = fmaf(-p, ex, 1.0f);                 // erf(z), z >= 0
    return fmaf(__builtin_copysignf(0.5f, t), e, 0.5f);
}

// ---------------------------------------------------------------------------
// Kernel 1: scores[b,m] = dot(x[b,m,:], w) + b0.  One wave (32 lanes) per row.
// ---------------------------------------------------------------------------
__global__ void scores_kernel(const float* __restrict__ x,
                              const float* __restrict__ w,
                              const float* __restrict__ bias,
                              float* __restrict__ scores) {
    int gt   = blockIdx.x * blockDim.x + threadIdx.x;
    int row  = gt >> 5;                 // 0 .. B*M-1
    int lane = threadIdx.x & 31;
    const float* xr = x + (size_t)row * C_;
    __builtin_prefetch(xr + 8 * C_, 0, 0);

    float4 a0 = *(const float4*)(xr + lane * 4);
    float4 a1 = *(const float4*)(xr + 128 + lane * 4);
    float4 w0 = *(const float4*)(w + lane * 4);
    float4 w1 = *(const float4*)(w + 128 + lane * 4);

    float s = 0.0f;
    s = fmaf(a0.x, w0.x, s); s = fmaf(a0.y, w0.y, s);
    s = fmaf(a0.z, w0.z, s); s = fmaf(a0.w, w0.w, s);
    s = fmaf(a1.x, w1.x, s); s = fmaf(a1.y, w1.y, s);
    s = fmaf(a1.z, w1.z, s); s = fmaf(a1.w, w1.w, s);

    #pragma unroll
    for (int off = 16; off > 0; off >>= 1) s += __shfl_xor(s, off, 32);

    if (lane == 0) scores[row] = s + bias[0];
}

// ---------------------------------------------------------------------------
// Kernel 2: flip/mask/clip -> s; build stable u64 sort keys for s_noisy.
// ---------------------------------------------------------------------------
__global__ void mask_key_kernel(const float* __restrict__ scores,
                                const float* __restrict__ u_mask,
                                const float* __restrict__ u_batch,
                                const float* __restrict__ rank_noise,
                                float* __restrict__ s_out,
                                unsigned long long* __restrict__ keys) {
    int t = blockIdx.x * blockDim.x + threadIdx.x;     // 0 .. B*M-1
    int b = t >> 12;                                   // /M_
    float sc = scores[t];
    float sr = scores[B_ * M_ - 1 - t];                // flip over (B,M) = flat reverse
    bool  mk = (u_mask[t] < 0.1f) && (u_batch[b] < 0.75f);
    float v  = mk ? fmaxf(sc, sr) : sc;
    float s  = fminf(fmaxf(v, -1.0f), 1.0f);
    s_out[t] = s;

    float sn = s + rank_noise[t] * STDDEV_F;
    unsigned u = __float_as_uint(sn);
    u = (u & 0x80000000u) ? ~u : (u | 0x80000000u);    // order-preserving map
    keys[t] = ((unsigned long long)u << 32) | (unsigned)(t & (M_ - 1));
}

// ---------------------------------------------------------------------------
// Kernel 3: stable ascending argsort per batch via bitonic sort on u64 keys
// in LDS (32 KB of the 320 KB/WGP). One 1024-thread block per batch.
// ---------------------------------------------------------------------------
__global__ __launch_bounds__(1024)
void sort_kernel(const unsigned long long* __restrict__ keys,
                 int* __restrict__ sorted_idx) {
    __shared__ unsigned long long k[M_];
    int b = blockIdx.x, tid = threadIdx.x;
    for (int i = tid; i < M_; i += 1024) k[i] = keys[b * M_ + i];
    __syncthreads();

    for (int size = 2; size <= M_; size <<= 1) {
        for (int stride = size >> 1; stride > 0; stride >>= 1) {
            for (int i = tid; i < M_; i += 1024) {
                int l = i ^ stride;
                if (l > i) {
                    bool up = ((i & size) == 0);
                    unsigned long long a = k[i], c = k[l];
                    if (up ? (a > c) : (a < c)) { k[i] = c; k[l] = a; }
                }
            }
            __syncthreads();
        }
    }
    for (int i = tid; i < M_; i += 1024)
        sorted_idx[b * M_ + i] = (int)(k[i] & 0xFFFFFFFFu);
}

// ---------------------------------------------------------------------------
// Kernel 4 (dominant): diff_ranks[b,i] = sum_j 0.5 + 0.5*erf((s_i-s_j)*inv).
// s[b,:] cached in LDS.  Each wave owns 16 i-rows; per iteration it builds a
// 16x32 f16 cdf tile in the A-layout (lane's 16 elements are two runs of 8
// consecutive j -> loaded as 4x ds_load_b128) and folds the row-sum into
// V_WMMA_F32_16X16X32_F16 against B=ones with f32 accumulation.  The erf is
// fully branch-free so the tile build stays a straight-line FMA/TRANS stream
// that co-executes with the XDL pipe.
// ---------------------------------------------------------------------------
__global__ __launch_bounds__(256)
void diff_ranks_kernel(const float* __restrict__ s,
                       float* __restrict__ dranks) {
    __shared__ float sl[M_];
    int b  = blockIdx.x >> 5;              // 32 blocks per batch, 128 rows each
    int ib = (blockIdx.x & 31) * 128;
    int tid = threadIdx.x;
    for (int i = tid; i < M_; i += 256) sl[i] = s[b * M_ + i];
    __syncthreads();

    int wv = tid >> 5, lane = tid & 31;
    int h = lane >> 4;                     // lane half selects K sub-range
    int r = lane & 15;                     // A-matrix row within tile
    int rowbase = ib + wv * 16;
    float si = sl[rowbase + r];

    v16h ones;
    #pragma unroll
    for (int e = 0; e < 16; ++e) ones[e] = (_Float16)1.0f;

    v8f acc = {};
    for (int jb = 0; jb < M_; jb += 32) {
        // lane's 16 A-elements: K = {8h..8h+7} U {16+8h..16+8h+7}
        const float4* r1 = (const float4*)(sl + jb + 8 * h);
        const float4* r2 = (const float4*)(sl + jb + 16 + 8 * h);
        float4 q0 = r1[0], q1 = r1[1];
        float4 q2 = r2[0], q3 = r2[1];
        float sj[16] = { q0.x, q0.y, q0.z, q0.w,  q1.x, q1.y, q1.z, q1.w,
                         q2.x, q2.y, q2.z, q2.w,  q3.x, q3.y, q3.z, q3.w };
        v16h a;
        #pragma unroll
        for (int e = 0; e < 16; ++e)
            a[e] = (_Float16)cdf_fast((si - sj[e]) * INV_F);

        // D = A(16x32 f16) x ones(32x16 f16) + C : every column = row sum
        acc = __builtin_amdgcn_wmma_f32_16x16x32_f16(
                  false, a, false, ones, (short)0, acc, false, false);
    }

    // D layout: lane L, vgpr d -> M = d + 8*(L/16), N = L%16 (all N equal)
    if ((lane & 15) == 0) {
        int base = b * M_ + rowbase + 8 * h;
        #pragma unroll
        for (int d = 0; d < 8; ++d) dranks[base + d] = acc[d];
    }
}

// ---------------------------------------------------------------------------
// Kernel 5: gather top-N rows, RMS, noise mix. One wave per output row.
// ---------------------------------------------------------------------------
__global__ void output_kernel(const float* __restrict__ x,
                              const float* __restrict__ out_noise,
                              const int* __restrict__ sorted_idx,
                              const float* __restrict__ dranks,
                              float* __restrict__ y_out,
                              int* __restrict__ idx_out,
                              float* __restrict__ xs_out) {
    int gw   = (blockIdx.x * blockDim.x + threadIdx.x) >> 5;  // 0 .. B*N-1
    int lane = threadIdx.x & 31;
    int b  = gw >> 11;                       // /N_
    int jj = gw & (N_ - 1);
    int pos = (M_ - N_) + jj;

    int   src = sorted_idx[b * M_ + pos];
    float dr  = dranks[b * M_ + src];
    float ranks = dr + ((float)pos - dr);    // forward == arange up to rounding
    float xs = fminf((ranks - (float)(M_ - N_)) / (0.1f * (float)N_), 1.0f);

    const float* xr = x + ((size_t)b * M_ + src) * C_;
    const float* nr = out_noise + ((size_t)b * N_ + jj) * C_;
    __builtin_prefetch(nr, 0, 0);

    float4 v0 = *(const float4*)(xr + lane * 4);
    float4 v1 = *(const float4*)(xr + 128 + lane * 4);
    float ss = v0.x*v0.x + v0.y*v0.y + v0.z*v0.z + v0.w*v0.w
             + v1.x*v1.x + v1.y*v1.y + v1.z*v1.z + v1.w*v1.w;
    #pragma unroll
    for (int off = 16; off > 0; off >>= 1) ss += __shfl_xor(ss, off, 32);

    float rms = sqrtf(ss * (1.0f / (float)C_) + 1e-5f);
    float ns  = (1.0f - xs) * rms;

    float4 n0 = *(const float4*)(nr + lane * 4);
    float4 n1 = *(const float4*)(nr + 128 + lane * 4);
    float4 o0, o1;
    o0.x = v0.x * xs + n0.x * ns;  o0.y = v0.y * xs + n0.y * ns;
    o0.z = v0.z * xs + n0.z * ns;  o0.w = v0.w * xs + n0.w * ns;
    o1.x = v1.x * xs + n1.x * ns;  o1.y = v1.y * xs + n1.y * ns;
    o1.z = v1.z * xs + n1.z * ns;  o1.w = v1.w * xs + n1.w * ns;

    float* yr = y_out + (size_t)gw * C_;
    *(float4*)(yr + lane * 4) = o0;
    *(float4*)(yr + 128 + lane * 4) = o1;

    if (lane == 0) { idx_out[gw] = src; xs_out[gw] = xs; }
}

// ---------------------------------------------------------------------------
extern "C" void kernel_launch(void* const* d_in, const int* in_sizes, int n_in,
                              void* d_out, int out_size, void* d_ws, size_t ws_size,
                              hipStream_t stream) {
    (void)in_sizes; (void)n_in; (void)out_size; (void)ws_size;
    const float* x          = (const float*)d_in[0];
    const float* w          = (const float*)d_in[1];
    const float* bias       = (const float*)d_in[2];
    const float* u_mask     = (const float*)d_in[3];
    const float* u_batch    = (const float*)d_in[4];
    const float* rank_noise = (const float*)d_in[5];
    const float* out_noise  = (const float*)d_in[6];

    char* ws = (char*)d_ws;                       // 384 KB total
    float*              scores  = (float*)(ws);                 //  64 KB
    float*              s_buf   = (float*)(ws + 65536);         //  64 KB
    unsigned long long* keys    = (unsigned long long*)(ws + 131072); // 128 KB
    int*                sidx    = (int*)(ws + 262144);          //  64 KB
    float*              dranks  = (float*)(ws + 327680);        //  64 KB

    int*   idx_out = (int*)d_out;                               // B*N ints
    float* y_out   = (float*)d_out + (B_ * N_);                 // B*N*C floats
    float* xs_out  = (float*)d_out + (B_ * N_) + (size_t)B_ * N_ * C_;

    scores_kernel    <<<(B_ * M_) / 8, 256, 0, stream>>>(x, w, bias, scores);
    mask_key_kernel  <<<(B_ * M_) / 256, 256, 0, stream>>>(scores, u_mask, u_batch,
                                                           rank_noise, s_buf, keys);
    sort_kernel      <<<B_, 1024, 0, stream>>>(keys, sidx);
    diff_ranks_kernel<<<(B_ * M_) / 128, 256, 0, stream>>>(s_buf, dranks);
    output_kernel    <<<(B_ * N_) / 8, 256, 0, stream>>>(x, out_noise, sidx, dranks,
                                                         y_out, idx_out, xs_out);
}
